// IgnitionMoE_2525440770145
// MI455X (gfx1250) — matrix operand order repaired and checked by reading, not compiled, not run on hardware
//
#include <hip/hip_runtime.h>

// ---------------------------------------------------------------------------
// Problem constants (from reference)
// ---------------------------------------------------------------------------
#define T_TOK   8192      // 4 * 2048 tokens
#define D_MODEL 1024
#define F_DIM   2048
#define N_EXP   8
#define LN_EPS  1e-5f

// B-tile staging: 64 columns x 128 K-halves per stage, padded rows in LDS.
#define KSTAGE     128              // bf16 elements of K per LDS stage
#define ROW_HALF   136              // 128 halves + 8 halves (16B) TDM pad -> bank-conflict break
// LDS per buffer: 64 rows * 272B = 17408B; double buffered.

// ---------------------------------------------------------------------------
// CDNA5 WMMA types
// ---------------------------------------------------------------------------
typedef __attribute__((ext_vector_type(16))) __bf16        v16bf;
typedef __attribute__((ext_vector_type(8)))  float         v8f;
typedef __attribute__((ext_vector_type(4)))  unsigned int  v4u;
typedef __attribute__((ext_vector_type(8)))  unsigned int  v8u;

union FragBF {
    v16bf bf;
    v4u   u[2];
};

__device__ inline unsigned short f32_to_bf16(float f) {
    unsigned int u = __float_as_uint(f);
    unsigned int r = u + 0x7FFFu + ((u >> 16) & 1u);   // round to nearest even
    return (unsigned short)(r >> 16);
}

// A fragment: 16x32 bf16, M = lane&15.
// lanes 0-15: halves 0..15 = K[0..15]
// lanes16-31: halves 0..15 = K[16..31]
__device__ inline v16bf load_a_frag(const unsigned short* __restrict__ rowPtr,
                                    int k0, int lane) {
    const unsigned short* p = rowPtr + k0 + ((lane & 16) ? 16 : 0);
    FragBF f;
    f.u[0] = *(const v4u*)(p);
    f.u[1] = *(const v4u*)(p + 8);
    return f.bf;
}

// B fragment from LDS: 32x16 bf16 tile, N = lane&15; B[k][n] = W[n][k].
// lanes 0-15 hold K=0..15, lanes 16-31 hold K=16..31 -> 16 contiguous halves.
__device__ inline v16bf load_b_frag_lds(const unsigned short* p) {
    FragBF f;
    f.u[0] = *(const v4u*)(p);
    f.u[1] = *(const v4u*)(p + 8);
    return f.bf;
}

// ---------------------------------------------------------------------------
// TDM: DMA one B stage (64 weight rows x 256B of K) from global into LDS.
// 2D descriptor, data_size = 8B elements:
//   tile_dim0 = 32 elems (256B/row), tile_dim1 = 64 rows,
//   tensor_dim0_stride = KDIM/4 (8B units between weight rows),
//   pad_enable: +4 DWORDs (16B) after every 64 DWORDs (one row) -> 272B pitch.
// Issued by wave 0 only; tracked with TENSORcnt.
// ---------------------------------------------------------------------------
__device__ inline void tdm_load_b_tile(const unsigned short* gp,
                                       unsigned ldsOff, unsigned strideElems) {
    unsigned long long ga = (unsigned long long)gp;
    v4u g0;
    g0[0] = 1u;                                            // count=1 (user descriptor)
    g0[1] = ldsOff;                                        // lds_addr (bytes)
    g0[2] = (unsigned)(ga & 0xFFFFFFFFull);                // global_addr[31:0]
    g0[3] = (unsigned)((ga >> 32) & 0x1FFFFFFull)          // global_addr[56:32]
          | (2u << 30);                                    // type = 2 ("image")
    v8u g1;
    g1[0] = (3u << 16)                                     // data_size = 8B
          | (1u << 20)                                     // pad_enable
          | (5u << 22)                                     // pad_interval = 64 DW
          | (3u << 25);                                    // pad_amount  = 4 DW (16B)
    g1[1] = (32u & 0xFFFFu) << 16;                         // tensor_dim0 = 32 (lo16)
    g1[2] = (64u & 0xFFFFu) << 16;                         // tensor_dim1 = 64 (lo16)
    g1[3] = (32u << 16);                                   // tile_dim0 = 32
    g1[4] = 64u;                                           // tile_dim1 = 64, tile_dim2 = 0
    g1[5] = strideElems;                                   // tensor_dim0_stride lo32
    g1[6] = 0u;
    g1[7] = 0u;
    asm volatile("tensor_load_to_lds %0, %1"
                 :: "s"(g0), "s"(g1)
                 : "memory");
}

#if __has_builtin(__builtin_amdgcn_s_wait_tensorcnt)
#define WAIT_TENSORCNT(n) __builtin_amdgcn_s_wait_tensorcnt(n)
#else
#define WAIT_TENSORCNT(n) asm volatile("s_wait_tensorcnt %0" :: "i"(n) : "memory")
#endif

// ---------------------------------------------------------------------------
// 1) BitLinear quantization: per output row, scale = max(mean|W|,1e-5),
//    Wq = clip(round(W/scale),-1,1)*scale  -> bf16.  One wave per row.
// ---------------------------------------------------------------------------
__global__ __launch_bounds__(256) void k_quant(const float* __restrict__ W,
                                               unsigned short* __restrict__ Wq,
                                               int cols) {
    int wave = threadIdx.x >> 5;
    int lane = threadIdx.x & 31;
    size_t row = (size_t)blockIdx.x * 8 + wave;
    const float* w = W + row * cols;
    float s = 0.f;
    for (int c = lane; c < cols; c += 32) s += fabsf(w[c]);
#pragma unroll
    for (int off = 16; off > 0; off >>= 1) s += __shfl_xor(s, off, 32);
    float scale = fmaxf(s / (float)cols, 1e-5f);
    float inv = 1.0f / scale;
    unsigned short* q = Wq + row * cols;
    for (int c = lane; c < cols; c += 32) {
        float r = rintf(w[c] * inv);
        r = fminf(fmaxf(r, -1.f), 1.f) * scale;
        q[c] = f32_to_bf16(r);
    }
}

// ---------------------------------------------------------------------------
// 2) Fused LayerNorm + router softmax/top-1.  One 256-thread block per token.
// ---------------------------------------------------------------------------
__global__ __launch_bounds__(256) void k_ln_router(const float* __restrict__ x,
                                                   const float* __restrict__ gamma,
                                                   const float* __restrict__ beta,
                                                   const float* __restrict__ rw,
                                                   unsigned short* __restrict__ Hbf,
                                                   float* __restrict__ gate,
                                                   int* __restrict__ topidx) {
    __shared__ float hs[D_MODEL];
    __shared__ float red[16];
    __shared__ float logits[N_EXP];

    int t = blockIdx.x;
    int tid = threadIdx.x;
    int wave = tid >> 5, lane = tid & 31;
    const float* xr = x + (size_t)t * D_MODEL;

    float v[4], sum = 0.f, sq = 0.f;
#pragma unroll
    for (int i = 0; i < 4; ++i) {
        v[i] = xr[tid + 256 * i];
        sum += v[i];
        sq  += v[i] * v[i];
    }
#pragma unroll
    for (int off = 16; off > 0; off >>= 1) {
        sum += __shfl_xor(sum, off, 32);
        sq  += __shfl_xor(sq,  off, 32);
    }
    if (lane == 0) { red[wave] = sum; red[8 + wave] = sq; }
    __syncthreads();
    if (tid == 0) {
        float S = 0.f, Q = 0.f;
        for (int i = 0; i < 8; ++i) { S += red[i]; Q += red[8 + i]; }
        float mu = S / (float)D_MODEL;
        float var = Q / (float)D_MODEL - mu * mu;
        red[0] = mu;
        red[1] = rsqrtf(var + LN_EPS);
    }
    __syncthreads();
    float mu = red[0], rs = red[1];
#pragma unroll
    for (int i = 0; i < 4; ++i) {
        int d = tid + 256 * i;
        float h = (v[i] - mu) * rs * gamma[d] + beta[d];
        hs[d] = h;
        Hbf[(size_t)t * D_MODEL + d] = f32_to_bf16(h);
    }
    __syncthreads();
    if (wave < N_EXP) {
        const float* r = rw + (size_t)wave * D_MODEL;
        float l = 0.f;
        for (int d = lane; d < D_MODEL; d += 32) l += hs[d] * r[d];
#pragma unroll
        for (int off = 16; off > 0; off >>= 1) l += __shfl_xor(l, off, 32);
        if (lane == 0) logits[wave] = l;
    }
    __syncthreads();
    if (tid == 0) {
        float mx = logits[0]; int mi = 0;
        for (int e = 1; e < N_EXP; ++e)
            if (logits[e] > mx) { mx = logits[e]; mi = e; }
        float den = 0.f;
        for (int e = 0; e < N_EXP; ++e) den += __expf(logits[e] - mx);
        gate[t] = 1.0f / den;     // softmax max prob = exp(0)/den
        topidx[t] = mi;
    }
}

// ---------------------------------------------------------------------------
// 3) Expert bucketing (top-1 routing)
// ---------------------------------------------------------------------------
__global__ void k_zero_counts(int* counts) {
    if (threadIdx.x < N_EXP) counts[threadIdx.x] = 0;
}

__global__ __launch_bounds__(256) void k_bucket(const int* __restrict__ topidx,
                                                int* __restrict__ counts,
                                                int* __restrict__ buckets) {
    int t = blockIdx.x * 256 + threadIdx.x;
    int e = topidx[t];
    int p = atomicAdd(&counts[e], 1);
    buckets[e * T_TOK + p] = t;
}

// ---------------------------------------------------------------------------
// 4) WMMA GEMM: C[M,N] = A[M,K] (bf16, token-row-indexed) x W[e][N,K]^T
//    Block = 128 threads (4 waves), tile 64x64. B staged in LDS via TDM,
//    double-buffered (TENSORcnt pipelined); A direct per-wave global b128.
//    EPI: 0 = SiLU -> bf16 store, 1 = f32 store, 2 = gated accumulate f32.
// ---------------------------------------------------------------------------
template <int KDIM, int EPI, bool GATHER>
__global__ __launch_bounds__(128) void k_gemm(const unsigned short* __restrict__ A,
                                              const unsigned short* __restrict__ W,
                                              unsigned short* __restrict__ Sout,
                                              float* __restrict__ Fout,
                                              const float* __restrict__ gate,
                                              const int* __restrict__ counts,
                                              const int* __restrict__ buckets,
                                              int N) {
    __shared__ unsigned short Bs[2][64 * ROW_HALF];
    __shared__ int tokS[64];

    const int e = blockIdx.z;
    const int mtile = blockIdx.x, ntile = blockIdx.y;
    int cnt = 0;
    if (GATHER) {
        cnt = counts[e];
        if (mtile * 64 >= cnt) return;
        if (threadIdx.x < 64) {
            int gi = mtile * 64 + (int)threadIdx.x;
            int gc = (gi < cnt) ? gi : (cnt - 1);
            tokS[threadIdx.x] = buckets[e * T_TOK + gc];
        }
        __syncthreads();
    }

    const int tid  = threadIdx.x;
    const int wave = tid >> 5, lane = tid & 31;

    const int rloc = wave * 16 + (lane & 15);
    const int tokA = GATHER ? tokS[rloc] : (mtile * 64 + rloc);
    const unsigned short* Arow = A + (size_t)tokA * KDIM;

    const int colBase = ntile * 64;
    const unsigned short* Wtile =
        W + (size_t)e * N * KDIM + (size_t)colBase * KDIM;

    // Raw LDS byte addresses of the two staging buffers for the TDM descriptor.
    const unsigned ldsB0 = (unsigned)(unsigned long long)
        (__attribute__((address_space(3))) unsigned short*)&Bs[0][0];
    const unsigned ldsB1 = (unsigned)(unsigned long long)
        (__attribute__((address_space(3))) unsigned short*)&Bs[1][0];

    constexpr int NSTAGE = KDIM / KSTAGE;
    const unsigned strideElems = (unsigned)(KDIM / 4);   // row stride in 8B units

    if (wave == 0)
        tdm_load_b_tile(Wtile, ldsB0, strideElems);      // prologue: stage 0

    const int nl = lane & 15;
    const int klane = (lane & 16) ? 16 : 0;

    v8f acc[4] = {};
#pragma unroll 1
    for (int s = 0; s < NSTAGE; ++s) {
        const int kbase = s * KSTAGE;
        if (s + 1 < NSTAGE && wave == 0)
            tdm_load_b_tile(Wtile + (size_t)(kbase + KSTAGE),
                            ((s + 1) & 1) ? ldsB1 : ldsB0, strideElems);
        if (s + 1 < NSTAGE) { WAIT_TENSORCNT(1); } else { WAIT_TENSORCNT(0); }
        __syncthreads();                                 // stage s visible to all waves

        const unsigned short* bufp = &Bs[s & 1][0];
        __builtin_prefetch(Arow + kbase + 2 * KSTAGE, 0, 1);   // global_prefetch_b8
#pragma unroll
        for (int kk = 0; kk < KSTAGE / 32; ++kk) {
            v16bf a = load_a_frag(Arow, kbase + kk * 32, lane);
#pragma unroll
            for (int j = 0; j < 4; ++j) {
                const unsigned short* bp =
                    bufp + (j * 16 + nl) * ROW_HALF + kk * 32 + klane;
                v16bf b = load_b_frag_lds(bp);
                acc[j] = __builtin_amdgcn_wmma_f32_16x16x32_bf16(
                    false, a, false, b, (short)0, acc[j], false, false);
            }
        }
        __syncthreads();                                 // all reads done before overwrite
    }

    // Epilogue. D layout: N = lane&15; VGPR i -> M = i (+8 for lanes 16-31).
#pragma unroll
    for (int j = 0; j < 4; ++j) {
        const int col = colBase + j * 16 + nl;
#pragma unroll
        for (int i = 0; i < 8; ++i) {
            const int br = wave * 16 + i + ((lane & 16) ? 8 : 0);
            const int tok = GATHER ? tokS[br] : (mtile * 64 + br);
            const bool valid = !GATHER || ((mtile * 64 + br) < cnt);
            const float vv = acc[j][i];
            if (EPI == 0) {
                if (valid) {
                    float sl = vv / (1.f + __expf(-vv));   // SiLU
                    Sout[(size_t)tok * N + col] = f32_to_bf16(sl);
                }
            } else if (EPI == 1) {
                Fout[(size_t)tok * N + col] = vv;
            } else {
                if (valid) {
                    Fout[(size_t)tok * N + col] += gate[tok] * vv;
                }
            }
        }
    }
}

// ---------------------------------------------------------------------------
// Host-side launcher
// ---------------------------------------------------------------------------
extern "C" void kernel_launch(void* const* d_in, const int* in_sizes, int n_in,
                              void* d_out, int out_size, void* d_ws, size_t ws_size,
                              hipStream_t stream) {
    const float* x     = (const float*)d_in[0];
    const float* sw1   = (const float*)d_in[1];
    const float* sw2   = (const float*)d_in[2];
    const float* ew1   = (const float*)d_in[3];
    const float* ew2   = (const float*)d_in[4];
    const float* rw    = (const float*)d_in[5];
    const float* gamma = (const float*)d_in[6];
    const float* beta  = (const float*)d_in[7];
    float* out = (float*)d_out;

    char* ws = (char*)d_ws;
    size_t off = 0;
    auto carve = [&](size_t bytes) -> void* {
        void* p = ws + off;
        off = (off + bytes + 255) & ~(size_t)255;
        return p;
    };

    unsigned short* Hbf  = (unsigned short*)carve((size_t)T_TOK * D_MODEL * 2);
    unsigned short* SW1q = (unsigned short*)carve((size_t)F_DIM * D_MODEL * 2);
    unsigned short* SW2q = (unsigned short*)carve((size_t)D_MODEL * F_DIM * 2);
    unsigned short* EW1q = (unsigned short*)carve((size_t)N_EXP * F_DIM * D_MODEL * 2);
    unsigned short* EW2q = (unsigned short*)carve((size_t)N_EXP * D_MODEL * F_DIM * 2);
    unsigned short* Smid = (unsigned short*)carve((size_t)T_TOK * F_DIM * 2);
    float* gateB = (float*)carve((size_t)T_TOK * 4);
    int*   idxB  = (int*)carve((size_t)T_TOK * 4);
    int*   cnts  = (int*)carve(256);
    int*   bkts  = (int*)carve((size_t)N_EXP * T_TOK * 4);
    (void)ws_size; (void)in_sizes; (void)n_in; (void)out_size;

    // 1) Quantize weights -> bf16 ternary
    k_quant<<<dim3(F_DIM / 8), 256, 0, stream>>>(sw1, SW1q, D_MODEL);
    k_quant<<<dim3(D_MODEL / 8), 256, 0, stream>>>(sw2, SW2q, F_DIM);
    k_quant<<<dim3(N_EXP * F_DIM / 8), 256, 0, stream>>>(ew1, EW1q, D_MODEL);
    k_quant<<<dim3(N_EXP * D_MODEL / 8), 256, 0, stream>>>(ew2, EW2q, F_DIM);

    // 2) LayerNorm + router
    k_ln_router<<<dim3(T_TOK), 256, 0, stream>>>(x, gamma, beta, rw, Hbf, gateB, idxB);

    // 3) Bucket tokens by expert
    k_zero_counts<<<dim3(1), 32, 0, stream>>>(cnts);
    k_bucket<<<dim3(T_TOK / 256), 256, 0, stream>>>(idxB, cnts, bkts);

    // 4) Shared expert: H x SW1^T -> SiLU -> bf16 S ; S x SW2^T -> f32 out
    k_gemm<D_MODEL, 0, false><<<dim3(T_TOK / 64, F_DIM / 64, 1), 128, 0, stream>>>(
        Hbf, SW1q, Smid, nullptr, nullptr, nullptr, nullptr, F_DIM);
    k_gemm<F_DIM, 1, false><<<dim3(T_TOK / 64, D_MODEL / 64, 1), 128, 0, stream>>>(
        Smid, SW2q, nullptr, out, nullptr, nullptr, nullptr, D_MODEL);

    // 5) Routed experts (token-gathered grouped GEMM, worst-case fixed grid)
    k_gemm<D_MODEL, 0, true><<<dim3(T_TOK / 64, F_DIM / 64, N_EXP), 128, 0, stream>>>(
        Hbf, EW1q, Smid, nullptr, nullptr, cnts, bkts, F_DIM);
    k_gemm<F_DIM, 2, true><<<dim3(T_TOK / 64, D_MODEL / 64, N_EXP), 128, 0, stream>>>(
        Smid, EW2q, nullptr, out, gateB, cnts, bkts, D_MODEL);
}
